// Attention_80642305950431
// MI455X (gfx1250) — matrix operand order repaired and checked by reading, not compiled, not run on hardware
//
#include <hip/hip_runtime.h>

// ---------------------------------------------------------------------------
// Problem constants (from reference): B=2, N=2048, D=1024, H=16, d=64
// ---------------------------------------------------------------------------
#define BATCH   2
#define SEQ     2048
#define EMB     1024
#define HEADS   16
#define HDIM    64
#define QKV_OUT (3 * HEADS * HDIM)   // 3072
#define MROWS   (BATCH * SEQ)        // 4096
#define SCALE_F 0.125f               // 64^-0.5
#define KSTEPS  (EMB / 32)           // 32
#define NTILES  (SEQ / 64)           // 32 key tiles in attention

typedef __bf16 v16bf __attribute__((ext_vector_type(16)));
typedef float  v8f   __attribute__((ext_vector_type(8)));
typedef unsigned int u32x4 __attribute__((ext_vector_type(4)));
typedef int    i32x4 __attribute__((ext_vector_type(4)));
typedef int    i32x8 __attribute__((ext_vector_type(8)));

union BF16Frag { v16bf v; unsigned int w[8]; unsigned short h[16]; };

__device__ __forceinline__ unsigned short f2bf(float f) {
    union { float f; unsigned int u; } c; c.f = f;
    unsigned int u = c.u;
    u += 0x7FFFu + ((u >> 16) & 1u);     // round-to-nearest-even
    return (unsigned short)(u >> 16);
}

// ---------------------------------------------------------------------------
// TDM: 2-D tile DMA global->LDS (bf16 elements), per CDNA5 ISA D# layout.
// ---------------------------------------------------------------------------
__device__ __forceinline__ void tdm_load_2d(void* lds_ptr, const void* gptr,
                                            unsigned tile_x, unsigned tile_y,
                                            unsigned tensor_x, unsigned tensor_y,
                                            unsigned stride_elems) {
    unsigned lds_addr = (unsigned)(unsigned long long)lds_ptr;  // low 32 = LDS byte addr
    unsigned long long ga = (unsigned long long)gptr;
    u32x4 g0;
    g0[0] = 1u;                                        // count=1, user descriptor
    g0[1] = lds_addr;                                  // lds_addr [63:32]
    g0[2] = (unsigned)(ga & 0xFFFFFFFFu);              // global_addr lo
    g0[3] = (unsigned)((ga >> 32) & 0x01FFFFFFu) | 0x80000000u; // addr hi + type=2
    i32x8 g1;
    g1[0] = 0x00010000;                                // data_size = 1 -> 2 bytes
    g1[1] = (int)((tensor_x & 0xFFFFu) << 16);         // tensor_dim0[15:0]
    g1[2] = (int)(((tensor_x >> 16) & 0xFFFFu) | ((tensor_y & 0xFFFFu) << 16));
    g1[3] = (int)(((tensor_y >> 16) & 0xFFFFu) | ((tile_x & 0xFFFFu) << 16));
    g1[4] = (int)(tile_y & 0xFFFFu);                   // tile_dim1 (tile_dim2 = 0)
    g1[5] = (int)stride_elems;                         // tensor_dim0_stride[31:0]
    g1[6] = 0;
    g1[7] = 0;
    i32x4 z4 = {0, 0, 0, 0};
#if __clang_major__ >= 23
    i32x8 z8 = {0, 0, 0, 0, 0, 0, 0, 0};
    __builtin_amdgcn_tensor_load_to_lds(g0, g1, z4, z4, z8, 0);
#else
    __builtin_amdgcn_tensor_load_to_lds(g0, g1, z4, z4, 0);
#endif
}

// ---------------------------------------------------------------------------
// Kernel 0a/0b: fp32 -> bf16 conversion (w also transposed to [n][k])
// ---------------------------------------------------------------------------
__global__ void cvt_x_kernel(const float* __restrict__ x,
                             unsigned short* __restrict__ xb, int n) {
    for (int i = blockIdx.x * blockDim.x + threadIdx.x; i < n;
         i += gridDim.x * blockDim.x)
        xb[i] = f2bf(x[i]);
}

__global__ void cvt_w_kernel(const float* __restrict__ w,
                             unsigned short* __restrict__ wbT) {
    const int n = EMB * QKV_OUT;
    for (int i = blockIdx.x * blockDim.x + threadIdx.x; i < n;
         i += gridDim.x * blockDim.x) {
        int col = i >> 10;            // output column n  (0..3071)
        int k   = i & 1023;           // reduction index  (0..1023)
        wbT[i] = f2bf(w[(size_t)k * QKV_OUT + col]);
    }
}

// ---------------------------------------------------------------------------
// Kernel 1: QKV GEMM, LDS-tiled with double-buffered TDM feed.
//   Workgroup (8 waves) computes 128(M) x 128(N); wave (wm,wn) owns 32x64.
//   K-loop: 32-wide steps; A(128x32) + B(128x32) bf16 tiles DMA'd into
//   ping-pong LDS buffers by wave 0 via tensor_load_to_lds.
// Epilogue scatters into the reference's "faithful reshape" Q/K layout
// ([B,H,N,64] bf16, Q pre-scaled) and V transposed ([B,H,64,N] bf16).
// ---------------------------------------------------------------------------
__global__ void __launch_bounds__(256)
qkv_gemm_kernel(const unsigned short* __restrict__ xb,
                const unsigned short* __restrict__ wbT,
                unsigned short* __restrict__ qb,
                unsigned short* __restrict__ kbuf,
                unsigned short* __restrict__ vtb) {
    __shared__ __align__(16) unsigned short a_lds[2][128 * 32];  // [row][k]
    __shared__ __align__(16) unsigned short b_lds[2][128 * 32];  // [col][k]

    const int wv   = threadIdx.x >> 5;
    const int lane = threadIdx.x & 31;
    const int wm   = wv & 3;                 // 4 M-slices of 32 rows
    const int wn   = wv >> 2;                // 2 N-slices of 64 cols
    const int bM   = blockIdx.x / (QKV_OUT / 128);   // 0..31
    const int bN   = blockIdx.x % (QKV_OUT / 128);   // 0..23
    const int tM0  = bM * 128;
    const int tN0  = bN * 128;

    v8f acc[2][4];
#pragma unroll
    for (int mi = 0; mi < 2; ++mi)
#pragma unroll
        for (int ni = 0; ni < 4; ++ni)
            acc[mi][ni] = (v8f){0.f, 0.f, 0.f, 0.f, 0.f, 0.f, 0.f, 0.f};

    const int abase = (lane < 16) ? 0 : 8;   // A K-packing
    const int bbase = (lane < 16) ? 0 : 8;   // B K-packing

    // Prologue: fill buffer 0
    if (wv == 0) {
        tdm_load_2d(&a_lds[0][0], xb  + (size_t)tM0 * EMB,
                    /*tile*/ 32, 128, /*tensor*/ EMB, MROWS,   /*stride*/ EMB);
        tdm_load_2d(&b_lds[0][0], wbT + (size_t)tN0 * EMB,
                    /*tile*/ 32, 128, /*tensor*/ EMB, QKV_OUT, /*stride*/ EMB);
    }

    for (int ks = 0; ks < KSTEPS; ++ks) {
        const int buf = ks & 1;
        if (wv == 0) {
            if (ks + 1 < KSTEPS) {      // prefetch next K-slice into other buffer
                const int k2 = (ks + 1) * 32;
                tdm_load_2d(&a_lds[buf ^ 1][0], xb  + (size_t)tM0 * EMB + k2,
                            32, 128, EMB, MROWS,   EMB);
                tdm_load_2d(&b_lds[buf ^ 1][0], wbT + (size_t)tN0 * EMB + k2,
                            32, 128, EMB, QKV_OUT, EMB);
                __builtin_amdgcn_s_wait_tensorcnt(2);  // oldest pair (buf) done
            } else {
                __builtin_amdgcn_s_wait_tensorcnt(0);
            }
        }
        __syncthreads();

        BF16Frag af[2];
#pragma unroll
        for (int mi = 0; mi < 2; ++mi) {
            const int rl = wm * 32 + mi * 16 + (lane & 15);
            *(uint4*)&af[mi].w[0] = *(const uint4*)&a_lds[buf][rl * 32 + abase];
            *(uint4*)&af[mi].w[4] = *(const uint4*)&a_lds[buf][rl * 32 + 16 + abase];
        }
#pragma unroll
        for (int ni = 0; ni < 4; ++ni) {
            BF16Frag bfr;
            const int cl = wn * 64 + ni * 16 + (lane & 15);
            *(uint4*)&bfr.w[0] = *(const uint4*)&b_lds[buf][cl * 32 + bbase];
            *(uint4*)&bfr.w[4] = *(const uint4*)&b_lds[buf][cl * 32 + bbase + 16];
#pragma unroll
            for (int mi = 0; mi < 2; ++mi)
                acc[mi][ni] = __builtin_amdgcn_wmma_f32_16x16x32_bf16(
                    false, af[mi].v, false, bfr.v, (short)0, acc[mi][ni],
                    false, false);
        }
        __syncthreads();   // all reads of buf done before it is refilled
    }

    // Epilogue: C/D layout -> faithful-reshape scatter
    const int mbase = (lane < 16) ? 0 : 8;
#pragma unroll
    for (int ni = 0; ni < 4; ++ni) {
        const int ocol = tN0 + wn * 64 + ni * 16 + (lane & 15);
        const int t  = ocol >> 10;          // 0=q 1=k 2=v
        const int cc = ocol & 1023;
#pragma unroll
        for (int mi = 0; mi < 2; ++mi) {
#pragma unroll
            for (int i = 0; i < 8; ++i) {
                const int orow = tM0 + wm * 32 + mi * 16 + mbase + i; // [B*N)
                const int b_   = orow >> 11;
                const int n_   = orow & 2047;
                const long f   = (long)n_ * 1024 + cc;
                const int h    = (int)(f >> 17);
                const int r    = (int)((f >> 6) & 2047);
                const int dd   = (int)(f & 63);
                const size_t bh = (size_t)b_ * HEADS + h;
                const float val = acc[mi][ni][i];
                if (t == 0)
                    qb[(bh * SEQ + r) * HDIM + dd] = f2bf(val * SCALE_F);
                else if (t == 1)
                    kbuf[(bh * SEQ + r) * HDIM + dd] = f2bf(val);
                else
                    vtb[(bh * HDIM + dd) * SEQ + r] = f2bf(val);
            }
        }
    }
}

// ---------------------------------------------------------------------------
// Kernel 2: flash attention. 4 waves / block, 64 query rows / block,
// 32 key-tiles of 64 keys streamed via double-buffered TDM into LDS.
// ---------------------------------------------------------------------------
__global__ void __launch_bounds__(128)
attn_kernel(const unsigned short* __restrict__ qb,
            const unsigned short* __restrict__ kb,
            const unsigned short* __restrict__ vtb,
            float* __restrict__ out) {
    __shared__ __align__(16) unsigned short k_lds[2][64 * 64];   // [key][d]
    __shared__ __align__(16) unsigned short vt_lds[2][64 * 64];  // [d][key]
    __shared__ __align__(16) float          S_lds[4][16][64];
    __shared__ __align__(16) unsigned short P_lds[4][16][64];
    __shared__ float m_lds[4][16];
    __shared__ float l_lds[4][16];
    __shared__ float a_lds[4][16];

    const int w    = threadIdx.x >> 5;
    const int lane = threadIdx.x & 31;
    const int bh   = blockIdx.x >> 5;          // 0..31
    const int r0   = (blockIdx.x & 31) << 6;   // query row block base

    if (lane < 16) { m_lds[w][lane] = -1e30f; l_lds[w][lane] = 0.f; }

    // Q A-fragments (kept in registers, already scaled by d^-1/2)
    BF16Frag aQ[2];
    {
        const unsigned short* qrow =
            qb + ((size_t)bh * SEQ + (r0 + w * 16 + (lane & 15))) * HDIM;
        const int ab = (lane < 16) ? 0 : 8;
#pragma unroll
        for (int s = 0; s < 2; ++s) {
            *(uint4*)&aQ[s].w[0] = *(const uint4*)(qrow + s * 32 + ab);
            *(uint4*)&aQ[s].w[4] = *(const uint4*)(qrow + s * 32 + 16 + ab);
        }
    }

    v8f o[4];
#pragma unroll
    for (int j = 0; j < 4; ++j)
        o[j] = (v8f){0.f, 0.f, 0.f, 0.f, 0.f, 0.f, 0.f, 0.f};

    const unsigned short* kbh  = kb  + (size_t)bh * SEQ * HDIM;
    const unsigned short* vtbh = vtb + (size_t)bh * HDIM * SEQ;

    // Prologue: fill buffer 0 with key tile 0
    if (w == 0) {
        tdm_load_2d(&k_lds[0][0],  kbh,  64, 64, 64,  SEQ, HDIM);
        tdm_load_2d(&vt_lds[0][0], vtbh, 64, 64, SEQ, 64,  SEQ);
    }
    __syncthreads();

    for (int t = 0; t < NTILES; ++t) {
        const int buf = t & 1;
        if (w == 0) {  // wave 0 drives the Tensor Data Mover
            if (t + 1 < NTILES) {
                const int c1 = (t + 1) << 6;
                tdm_load_2d(&k_lds[buf ^ 1][0],  kbh + (size_t)c1 * HDIM,
                            64, 64, 64,  SEQ, HDIM);
                tdm_load_2d(&vt_lds[buf ^ 1][0], vtbh + c1,
                            64, 64, SEQ, 64,  SEQ);
                __builtin_amdgcn_s_wait_tensorcnt(2);  // current tile landed
            } else {
                __builtin_amdgcn_s_wait_tensorcnt(0);
            }
        }
        __syncthreads();

        // ---- S = (Q * d^-1/2) K^T  (fp32 accum) -> S_lds ----
#pragma unroll
        for (int j = 0; j < 4; ++j) {
            v8f s = {0.f, 0.f, 0.f, 0.f, 0.f, 0.f, 0.f, 0.f};
            const int key = j * 16 + (lane & 15);
#pragma unroll
            for (int ks = 0; ks < 2; ++ks) {
                BF16Frag bK;
                const int kd = ks * 32 + ((lane < 16) ? 0 : 8);
                *(uint4*)&bK.w[0] = *(const uint4*)&k_lds[buf][key * 64 + kd];
                *(uint4*)&bK.w[4] = *(const uint4*)&k_lds[buf][key * 64 + kd + 16];
                s = __builtin_amdgcn_wmma_f32_16x16x32_bf16(
                    false, aQ[ks].v, false, bK.v, (short)0, s, false, false);
            }
            const int rbase = (lane < 16) ? 0 : 8;
#pragma unroll
            for (int i = 0; i < 8; ++i)
                S_lds[w][rbase + i][j * 16 + (lane & 15)] = s[i];
        }
        __syncthreads();

        // ---- online softmax (one row per lane 0..15) ----
        if (lane < 16) {
            const int row = lane;
            float mx = -1e30f;
#pragma unroll
            for (int c = 0; c < 64; c += 4) {
                float4 v = *(const float4*)&S_lds[w][row][c];
                mx = fmaxf(mx, fmaxf(fmaxf(v.x, v.y), fmaxf(v.z, v.w)));
            }
            const float mold  = m_lds[w][row];
            const float mnew  = fmaxf(mold, mx);
            const float alpha = __expf(mold - mnew);
            float sum = 0.f;
#pragma unroll
            for (int c = 0; c < 64; c += 4) {
                float4 v = *(const float4*)&S_lds[w][row][c];
                const float p0 = __expf(v.x - mnew), p1 = __expf(v.y - mnew);
                const float p2 = __expf(v.z - mnew), p3 = __expf(v.w - mnew);
                sum += (p0 + p1) + (p2 + p3);
                P_lds[w][row][c + 0] = f2bf(p0);
                P_lds[w][row][c + 1] = f2bf(p1);
                P_lds[w][row][c + 2] = f2bf(p2);
                P_lds[w][row][c + 3] = f2bf(p3);
            }
            l_lds[w][row] = l_lds[w][row] * alpha + sum;
            m_lds[w][row] = mnew;
            a_lds[w][row] = alpha;
        }
        __syncthreads();

        // ---- rescale O and accumulate O += P V ----
        {
            const int rbase = (lane < 16) ? 0 : 8;
            float av[8];
#pragma unroll
            for (int i = 0; i < 8; ++i) av[i] = a_lds[w][rbase + i];
#pragma unroll
            for (int j = 0; j < 4; ++j)
#pragma unroll
                for (int i = 0; i < 8; ++i) o[j][i] *= av[i];

            BF16Frag aP[2];
            const int prow = lane & 15;
            const int ab = (lane < 16) ? 0 : 8;
#pragma unroll
            for (int s = 0; s < 2; ++s) {
                *(uint4*)&aP[s].w[0] = *(const uint4*)&P_lds[w][prow][s * 32 + ab];
                *(uint4*)&aP[s].w[4] = *(const uint4*)&P_lds[w][prow][s * 32 + 16 + ab];
            }
#pragma unroll
            for (int j = 0; j < 4; ++j) {
                const int dd = j * 16 + (lane & 15);
#pragma unroll
                for (int s = 0; s < 2; ++s) {
                    BF16Frag bV;
                    const int kk = s * 32 + ((lane < 16) ? 0 : 8);
                    *(uint4*)&bV.w[0] = *(const uint4*)&vt_lds[buf][dd * 64 + kk];
                    *(uint4*)&bV.w[4] = *(const uint4*)&vt_lds[buf][dd * 64 + kk + 16];
                    o[j] = __builtin_amdgcn_wmma_f32_16x16x32_bf16(
                        false, aP[s].v, false, bV.v, (short)0, o[j], false, false);
                }
            }
        }
        __syncthreads();   // reads of buf done before next refill
    }

    // ---- normalize and store (output layout == flat [B,H,N,d]) ----
    {
        const int rbase = (lane < 16) ? 0 : 8;
        float li[8];
#pragma unroll
        for (int i = 0; i < 8; ++i) li[i] = 1.0f / l_lds[w][rbase + i];
#pragma unroll
        for (int j = 0; j < 4; ++j) {
            const int dd = j * 16 + (lane & 15);
#pragma unroll
            for (int i = 0; i < 8; ++i) {
                const int row = r0 + w * 16 + rbase + i;
                out[((size_t)bh * SEQ + row) * HDIM + dd] = o[j][i] * li[i];
            }
        }
    }
}

// ---------------------------------------------------------------------------
// Host launcher
// ---------------------------------------------------------------------------
extern "C" void kernel_launch(void* const* d_in, const int* in_sizes, int n_in,
                              void* d_out, int out_size, void* d_ws, size_t ws_size,
                              hipStream_t stream) {
    (void)in_sizes; (void)n_in; (void)out_size; (void)ws_size;
    const float* x = (const float*)d_in[0];
    const float* w = (const float*)d_in[1];
    float* out = (float*)d_out;

    // Workspace partition (bf16 = unsigned short), total ~38 MB
    unsigned short* xb  = (unsigned short*)d_ws;                 // [4096][1024]
    unsigned short* wbT = xb  + (size_t)MROWS * EMB;             // [3072][1024]
    unsigned short* qb  = wbT + (size_t)QKV_OUT * EMB;           // [32][2048][64]
    unsigned short* kb  = qb  + (size_t)BATCH * HEADS * SEQ * HDIM;
    unsigned short* vtb = kb  + (size_t)BATCH * HEADS * SEQ * HDIM; // [32][64][2048]

    cvt_x_kernel<<<2048, 256, 0, stream>>>(x, xb, MROWS * EMB);
    cvt_w_kernel<<<2048, 256, 0, stream>>>(w, wbT);

    // (4096/128) * (3072/128) = 32*24 = 768 blocks, 8 waves each
    qkv_gemm_kernel<<<(MROWS / 128) * (QKV_OUT / 128), 256, 0, stream>>>(
        xb, wbT, qb, kb, vtb);

    // B*H * (N/64) = 32 * 32 = 1024 blocks, 128 threads (4 waves)
    attn_kernel<<<BATCH * HEADS * (SEQ / 64), 128, 0, stream>>>(qb, kb, vtb, out);
}
